// Attention_5239860101695
// MI455X (gfx1250) — matrix operand order repaired and checked by reading, not compiled, not run on hardware
//
#include <hip/hip_runtime.h>
#include <hip/hip_bf16.h>
#include <math.h>
#include <stdint.h>

// Problem constants (from reference): B=2, N=2048, DIM=1024, HEADS=16, HEAD_DIM=64
#define B_    2
#define N_    2048
#define C_    1024
#define H_    16
#define D_    64
#define K3_   3072          // 3*DIM
#define EPS_  1e-5f
#define SCALE_ 0.125f       // HEAD_DIM^-0.5

typedef _Float16 half4 __attribute__((ext_vector_type(4)));
typedef _Float16 half8 __attribute__((ext_vector_type(8)));
typedef _Float16 v16h  __attribute__((ext_vector_type(16)));
typedef float    v4f   __attribute__((ext_vector_type(4)));
typedef float    v8f   __attribute__((ext_vector_type(8)));
typedef unsigned int u32x4 __attribute__((ext_vector_type(4)));
typedef int          i32x4 __attribute__((ext_vector_type(4)));
typedef int          i32x8 __attribute__((ext_vector_type(8)));

union AFrag { v16h v; half8 h[2]; };

// Attention LDS layout
#define SP_   (N_ + 8)                    // fp32 S row pitch (bank-conflict pad)
#define PP_   (N_ + 16)                   // f16 P row pitch (keeps 16B alignment)
#define SMEM_S_BYTES   (16 * SP_ * 4)
#define SMEM_P_BYTES   (16 * PP_ * 2)
#define KTILE_BYTES    (16 * D_ * 2)      // 2 KB : 16 keys x 64 d
#define VTILE_BYTES    (D_ * 32 * 2)      // 4 KB : 64 d x 32 keys
#define ATTN_SMEM      (SMEM_S_BYTES + SMEM_P_BYTES + 2 * KTILE_BYTES + 2 * VTILE_BYTES)

#if __has_builtin(__builtin_amdgcn_tensor_load_to_lds) && \
    __has_builtin(__builtin_amdgcn_s_wait_tensorcnt)
#define HAVE_TDM 1
#endif

__device__ __forceinline__ v8f wmma_f16(v16h a, v16h b, v8f c) {
  // 8 args: (neg_a, A, neg_b, B, c_mod, C, reuse_a, reuse_b)
  return __builtin_amdgcn_wmma_f32_16x16x32_f16(false, a, false, b, (short)0, c,
                                                false, false);
}

#ifdef HAVE_TDM
// Tensor Data Mover: 2D tile load, descriptor per CDNA5 ISA D# layout (§8).
//   group0: [1:0]=count=1, [63:32]=lds_addr, [120:64]=global_addr, [127:126]=type=2
//   group1: [17:16]=data_size=1 (2B), [79:48]=tensor_dim0, [111:80]=tensor_dim1,
//           [127:112]=tile_dim0, [143:128]=tile_dim1, [207:160]=tensor_dim0_stride
__device__ __forceinline__ void tdm_load_2d(uint64_t ga, unsigned lds_off,
                                            unsigned td0, unsigned td1,
                                            unsigned stride0,
                                            unsigned tile0, unsigned tile1) {
  u32x4 g0 = { 1u,
               lds_off,
               (unsigned)(ga & 0xffffffffu),
               (unsigned)((ga >> 32) & 0x01ffffffu) | (2u << 30) };
  i32x8 g1 = { (int)(1u << 16),                                   // data_size=2B
               (int)((td0 & 0xffffu) << 16),
               (int)(((td0 >> 16) & 0xffffu) | ((td1 & 0xffffu) << 16)),
               (int)(((td1 >> 16) & 0xffffu) | ((tile0 & 0xffffu) << 16)),
               (int)(tile1 & 0xffffu),                            // tile_dim2 = 0
               (int)stride0, 0, 0 };
  i32x4 z4 = {0, 0, 0, 0};
#if __clang_major__ >= 23
  i32x8 z8 = {0, 0, 0, 0, 0, 0, 0, 0};
  __builtin_amdgcn_tensor_load_to_lds(g0, g1, z4, z4, z8, 0);
#else
  __builtin_amdgcn_tensor_load_to_lds(g0, g1, z4, z4, 0);
#endif
}
#endif  // HAVE_TDM

// ---------------------------------------------------------------------------
// f32 -> f16 elementwise convert
// ---------------------------------------------------------------------------
__global__ __launch_bounds__(256) void cvt_f32_f16(const float* __restrict__ src,
                                                   _Float16* __restrict__ dst, int n) {
  int i = blockIdx.x * blockDim.x + threadIdx.x;
  if (i < n) dst[i] = (_Float16)src[i];
}

// ---------------------------------------------------------------------------
// GEMM: C[M,Ncol] = A[M,K] * Bw[Ncol,K]^T (+ bias), f16 inputs, fp32 out.
// One wave computes a 64x64 tile (16 WMMA accumulators); fragments loaded
// directly from global (tiles stay L2-resident) with contiguous k per lane half (k = 16*hf + s).
// ---------------------------------------------------------------------------
__global__ __launch_bounds__(32) void gemm_nt_wmma(const _Float16* __restrict__ A,
                                                   const _Float16* __restrict__ Bw,
                                                   const float* __restrict__ bias,
                                                   float* __restrict__ Cout,
                                                   int M, int Ncol, int K) {
  const int lane = threadIdx.x & 31;
  const int l16  = lane & 15;
  const int hf   = lane >> 4;
  const int rm   = blockIdx.x * 64;
  const int cn   = blockIdx.y * 64;

  v8f acc[4][4];
#pragma unroll
  for (int i = 0; i < 4; ++i)
#pragma unroll
    for (int j = 0; j < 4; ++j) acc[i][j] = (v8f){};

  for (int k0 = 0; k0 < K; k0 += 32) {
    AFrag a[4];
    v16h  b[4];
#pragma unroll
    for (int i = 0; i < 4; ++i) {
      const _Float16* ap = A + (size_t)(rm + i * 16 + l16) * K + k0 + hf * 16;
      a[i].h[0] = *(const half8*)ap;
      a[i].h[1] = *(const half8*)(ap + 8);
    }
#pragma unroll
    for (int j = 0; j < 4; ++j) {
      const _Float16* bp = Bw + (size_t)(cn + j * 16 + l16) * K + k0 + hf * 16;
      b[j] = *(const v16h*)bp;
    }
#pragma unroll
    for (int i = 0; i < 4; ++i)
#pragma unroll
      for (int j = 0; j < 4; ++j) acc[i][j] = wmma_f16(a[i].v, b[j], acc[i][j]);
  }

  // C/D layout: VGPR r, lane L -> (M = r + 8*(L/16), N = L%16)
#pragma unroll
  for (int i = 0; i < 4; ++i)
#pragma unroll
    for (int j = 0; j < 4; ++j)
#pragma unroll
      for (int r = 0; r < 8; ++r) {
        int row = rm + i * 16 + r + 8 * hf;
        int col = cn + j * 16 + l16;
        float v = acc[i][j][r];
        if (bias) v += bias[col];
        Cout[(size_t)row * Ncol + col] = v;
      }
}

// ---------------------------------------------------------------------------
// Per-head layernorm of q,k (+ q*scale), pack to f16; v packed transposed.
//   qkv32 : [B*N, 3072] fp32, col = which*1024 + h*64 + d
//   qh,kh : [B,H,N,D] f16 ; vhT : [B,H,D,N] f16
// ---------------------------------------------------------------------------
__global__ __launch_bounds__(256) void ln_pack_kernel(const float* __restrict__ qkv,
                                                      const float* __restrict__ qnw,
                                                      const float* __restrict__ qnb,
                                                      const float* __restrict__ knw,
                                                      const float* __restrict__ knb,
                                                      _Float16* __restrict__ qh,
                                                      _Float16* __restrict__ kh,
                                                      _Float16* __restrict__ vhT) {
  int r = blockIdx.x * blockDim.x + threadIdx.x;
  if (r >= B_ * H_ * N_) return;
  int n = r % N_;
  int h = (r / N_) % H_;
  int b = r / (N_ * H_);
  const size_t rowbase = ((size_t)(b * N_ + n)) * K3_ + (size_t)h * D_;
  const size_t obase   = (((size_t)(b * H_ + h)) * N_ + n) * D_;

  for (int which = 0; which < 2; ++which) {
    const float* g = qkv + rowbase + which * C_;
    float mu = 0.f;
    for (int d = 0; d < D_; ++d) mu += g[d];
    mu *= (1.f / D_);
    float var = 0.f;
    for (int d = 0; d < D_; ++d) { float t = g[d] - mu; var += t * t; }
    var *= (1.f / D_);
    float inv = rsqrtf(var + EPS_);
    const float* wv = which ? knw : qnw;
    const float* bb = which ? knb : qnb;
    _Float16*   dst = which ? kh  : qh;
    float sc = which ? 1.f : SCALE_;
    for (int d = 0; d < D_; ++d)
      dst[obase + d] = (_Float16)((((g[d] - mu) * inv) * wv[d] + bb[d]) * sc);
  }
  const float* gv = qkv + rowbase + 2 * C_;
  const size_t vb = ((size_t)(b * H_ + h)) * D_ * N_;
  for (int d = 0; d < D_; ++d)
    vhT[vb + (size_t)d * N_ + n] = (_Float16)gv[d];
}

// ---------------------------------------------------------------------------
// Attention: one wave per 16 query rows.
//   K tiles (16x64) and V^T tiles (64x32) streamed into LDS double-buffers by
//   the Tensor Data Mover (s_wait_tensorcnt sync); S kept fp32 in LDS; row
//   softmax via wave32 cross-lane shuffles; P@V via WMMA with LDS A-fragments.
// ---------------------------------------------------------------------------
__global__ __launch_bounds__(32) void attn_kernel(const _Float16* __restrict__ qh,
                                                  const _Float16* __restrict__ kh,
                                                  const _Float16* __restrict__ vhT,
                                                  const unsigned char* __restrict__ mask,
                                                  _Float16* __restrict__ avh) {
  const int lane = threadIdx.x & 31;
  const int l16  = lane & 15;
  const int hf   = lane >> 4;
  const int qt = blockIdx.x, h = blockIdx.y, b = blockIdx.z;
  const int q0 = qt * 16;

  extern __shared__ char smem[];
  float*    S = (float*)smem;
  _Float16* P = (_Float16*)(smem + SMEM_S_BYTES);
  char* kbase = smem + SMEM_S_BYTES + SMEM_P_BYTES;
  char* vbase = kbase + 2 * KTILE_BYTES;

  const size_t bh = (size_t)(b * H_ + h);
  const _Float16* qh_ = qh  + bh * N_ * D_;
  const _Float16* kh_ = kh  + bh * N_ * D_;
  const _Float16* vt_ = vhT + bh * D_ * N_;
  const unsigned char* mrow = mask + (size_t)b * N_ * N_;  // mask[b,0,q,key] (bool bytes)

  // Q fragments (row = q0+l16, k = c*32 + 16*hf + s, contiguous per lane half)
  AFrag qa[2];
#pragma unroll
  for (int c = 0; c < 2; ++c) {
    const _Float16* p = qh_ + (size_t)(q0 + l16) * D_ + c * 32 + hf * 16;
    qa[c].h[0] = *(const half8*)p;
    qa[c].h[1] = *(const half8*)(p + 8);
  }

  // ---- S = Q K^T over all key tiles ----
  constexpr int NT = N_ / 16;
#ifdef HAVE_TDM
  {
    const unsigned klds = (unsigned)(uintptr_t)kbase;
    const uint64_t kga  = (uint64_t)(uintptr_t)kh_;
    // prologue: tile 0
    tdm_load_2d(kga, klds, D_, N_, D_, D_, 16);
    for (int j = 0; j < NT; ++j) {
      if (j + 1 < NT) {
        tdm_load_2d(kga + (uint64_t)(j + 1) * 16 * D_ * 2,
                    klds + (unsigned)((j + 1) & 1) * KTILE_BYTES,
                    D_, N_, D_, D_, 16);
        __builtin_amdgcn_s_wait_tensorcnt(1);   // tile j landed (in-order)
      } else {
        __builtin_amdgcn_s_wait_tensorcnt(0);
      }
      asm volatile("" ::: "memory");
      const _Float16* kt = (const _Float16*)(kbase + (j & 1) * KTILE_BYTES);
      v16h kb0 = *(const v16h*)(kt + l16 * D_ + 0  + hf * 16);
      v16h kb1 = *(const v16h*)(kt + l16 * D_ + 32 + hf * 16);
      v8f s = (v8f){};
      s = wmma_f16(qa[0].v, kb0, s);
      s = wmma_f16(qa[1].v, kb1, s);
      const int key = j * 16 + l16;
#pragma unroll
      for (int r = 0; r < 8; ++r) {
        int m = r + 8 * hf;
        float v = s[r];
        if (mrow[(size_t)(q0 + m) * N_ + key]) v = -INFINITY;
        S[m * SP_ + key] = v;
      }
    }
  }
#else
  for (int j = 0; j < NT; ++j) {
    const _Float16* kp = kh_ + (size_t)(j * 16 + l16) * D_;
    v16h kb0 = *(const v16h*)(kp + 0  + hf * 16);
    v16h kb1 = *(const v16h*)(kp + 32 + hf * 16);
    v8f s = (v8f){};
    s = wmma_f16(qa[0].v, kb0, s);
    s = wmma_f16(qa[1].v, kb1, s);
    const int key = j * 16 + l16;
#pragma unroll
    for (int r = 0; r < 8; ++r) {
      int m = r + 8 * hf;
      float v = s[r];
      if (mrow[(size_t)(q0 + m) * N_ + key]) v = -INFINITY;
      S[m * SP_ + key] = v;
    }
  }
#endif
  __syncthreads();

  // ---- row softmax: lane l16 owns row l16; halves split across hf ----
  {
    const int row = l16;
    float* base = S + row * SP_ + hf * (N_ / 2);   // 16B aligned (SP_, N_/2 mult of 4)
    const v4f* b4 = (const v4f*)base;
    float mx = -INFINITY;
    for (int c = 0; c < N_ / 8; ++c) {
      v4f t = b4[c];
      mx = fmaxf(mx, fmaxf(fmaxf(t[0], t[1]), fmaxf(t[2], t[3])));
    }
    mx = fmaxf(mx, __shfl_xor(mx, 16, 32));
    const bool dead = (mx == -INFINITY);
    float sum = 0.f;
    for (int c = 0; c < N_ / 8; ++c) {
      v4f t = b4[c];
      v4f e;
#pragma unroll
      for (int u = 0; u < 4; ++u) e[u] = dead ? 0.f : __expf(t[u] - mx);
      ((v4f*)base)[c] = e;
      sum += e[0] + e[1] + e[2] + e[3];
    }
    sum += __shfl_xor(sum, 16, 32);
    float inv = (sum > 0.f) ? (1.f / sum) : 0.f;
    half4* p4 = (half4*)(P + row * PP_ + hf * (N_ / 2));
    for (int c = 0; c < N_ / 8; ++c) {
      v4f t = b4[c];
      half4 o;
#pragma unroll
      for (int u = 0; u < 4; ++u) o[u] = (_Float16)(t[u] * inv);
      p4[c] = o;
    }
  }
  __syncthreads();

  // ---- O = P V : A-fragments from LDS P, V tiles streamed via TDM ----
  v8f o[4];
#pragma unroll
  for (int z = 0; z < 4; ++z) o[z] = (v8f){};
  constexpr int NT2 = N_ / 32;
#ifdef HAVE_TDM
  {
    const unsigned vlds = (unsigned)(uintptr_t)vbase;
    const uint64_t vga  = (uint64_t)(uintptr_t)vt_;
    tdm_load_2d(vga, vlds, N_, D_, N_, 32, D_);   // rows d=0..63, cols k0..k0+31
    for (int t = 0; t < NT2; ++t) {
      if (t + 1 < NT2) {
        tdm_load_2d(vga + (uint64_t)(t + 1) * 32 * 2,
                    vlds + (unsigned)((t + 1) & 1) * VTILE_BYTES,
                    N_, D_, N_, 32, D_);
        __builtin_amdgcn_s_wait_tensorcnt(1);
      } else {
        __builtin_amdgcn_s_wait_tensorcnt(0);
      }
      asm volatile("" ::: "memory");
      const int k0 = t * 32;
      AFrag pa;
      const _Float16* pp = P + (size_t)l16 * PP_ + k0 + hf * 16;
      pa.h[0] = *(const half8*)pp;
      pa.h[1] = *(const half8*)(pp + 8);
      const _Float16* vtile = (const _Float16*)(vbase + (t & 1) * VTILE_BYTES);
#pragma unroll
      for (int dblk = 0; dblk < 4; ++dblk) {
        v16h vb = *(const v16h*)(vtile + (dblk * 16 + l16) * 32 + hf * 16);
        o[dblk] = wmma_f16(pa.v, vb, o[dblk]);
      }
    }
  }
#else
  for (int k0 = 0; k0 < N_; k0 += 32) {
    AFrag pa;
    const _Float16* pp = P + (size_t)l16 * PP_ + k0 + hf * 16;
    pa.h[0] = *(const half8*)pp;
    pa.h[1] = *(const half8*)(pp + 8);
#pragma unroll
    for (int dblk = 0; dblk < 4; ++dblk) {
      v16h vb = *(const v16h*)(vt_ + (size_t)(dblk * 16 + l16) * N_ + k0 + hf * 16);
      o[dblk] = wmma_f16(pa.v, vb, o[dblk]);
    }
  }
#endif

#pragma unroll
  for (int dblk = 0; dblk < 4; ++dblk)
#pragma unroll
    for (int r = 0; r < 8; ++r) {
      int q = q0 + r + 8 * hf;
      int d = dblk * 16 + l16;
      avh[((size_t)(b * N_ + q)) * C_ + h * D_ + d] = (_Float16)o[dblk][r];
    }
}

// ---------------------------------------------------------------------------
extern "C" void kernel_launch(void* const* d_in, const int* in_sizes, int n_in,
                              void* d_out, int out_size, void* d_ws, size_t ws_size,
                              hipStream_t stream) {
  const float* x      = (const float*)d_in[0];
  const unsigned char* mask = (const unsigned char*)d_in[1];  // jnp bool -> 1 byte
  const float* qkv_w  = (const float*)d_in[2];
  const float* qnw    = (const float*)d_in[3];
  const float* qnb    = (const float*)d_in[4];
  const float* knw    = (const float*)d_in[5];
  const float* knb    = (const float*)d_in[6];
  const float* proj_w = (const float*)d_in[7];
  const float* proj_b = (const float*)d_in[8];
  float* out = (float*)d_out;

  char* wptr = (char*)d_ws;
  auto alloc = [&](size_t bytes) -> char* {
    char* p = wptr;
    wptr += (bytes + 255) & ~(size_t)255;
    return p;
  };
  const size_t MN = (size_t)B_ * N_;          // 4096 rows
  _Float16* xh     = (_Float16*)alloc(MN * C_ * 2);
  _Float16* wqkvh  = (_Float16*)alloc((size_t)K3_ * C_ * 2);
  _Float16* wprojh = (_Float16*)alloc((size_t)C_ * C_ * 2);
  float*    qkv32  = (float*)   alloc(MN * K3_ * 4);
  _Float16* qh     = (_Float16*)alloc((size_t)B_ * H_ * N_ * D_ * 2);
  _Float16* kh     = (_Float16*)alloc((size_t)B_ * H_ * N_ * D_ * 2);
  _Float16* vhT    = (_Float16*)alloc((size_t)B_ * H_ * D_ * N_ * 2);
  _Float16* avh    = (_Float16*)alloc(MN * C_ * 2);

  // 1) convert fp32 -> f16
  {
    int n1 = (int)(MN * C_);
    cvt_f32_f16<<<(n1 + 255) / 256, 256, 0, stream>>>(x, xh, n1);
    int n2 = K3_ * C_;
    cvt_f32_f16<<<(n2 + 255) / 256, 256, 0, stream>>>(qkv_w, wqkvh, n2);
    int n3 = C_ * C_;
    cvt_f32_f16<<<(n3 + 255) / 256, 256, 0, stream>>>(proj_w, wprojh, n3);
  }

  // 2) QKV = x @ qkv_w^T  (4096 x 3072 x 1024)
  gemm_nt_wmma<<<dim3((unsigned)(MN / 64), K3_ / 64), 32, 0, stream>>>(
      xh, wqkvh, nullptr, qkv32, (int)MN, K3_, C_);

  // 3) layernorm q,k (+scale) and pack q/k/v as f16 (v transposed)
  {
    int rows = B_ * H_ * N_;
    ln_pack_kernel<<<(rows + 255) / 256, 256, 0, stream>>>(
        qkv32, qnw, qnb, knw, knb, qh, kh, vhT);
  }

  // 4) masked softmax attention (TDM-streamed K/V)
  attn_kernel<<<dim3(N_ / 16, H_, B_), 32, ATTN_SMEM, stream>>>(
      qh, kh, vhT, mask, avh);

  // 5) out = attn_out @ proj_w^T + proj_b  (4096 x 1024 x 1024)
  gemm_nt_wmma<<<dim3((unsigned)(MN / 64), C_ / 64), 32, 0, stream>>>(
      avh, wprojh, proj_b, out, (int)MN, C_, C_);
}
